// SDMStore_61538291417811
// MI455X (gfx1250) — matrix operand order repaired and checked by reading, not compiled, not run on hardware
//
#include <hip/hip_runtime.h>

#define NTOK 2048
#define DDIM 2048
#define IDIM 8192
#define KMAX 64
#define KC   64                 // K elements per staged chunk
#define AROWS 128
#define BROWS 256
#define RSTR 72                 // padded LDS row stride in f16 (144 B)

typedef __attribute__((ext_vector_type(16))) _Float16 v16h;
typedef __attribute__((ext_vector_type(8)))  float    v8f;
typedef unsigned int u32x4 __attribute__((ext_vector_type(4)));
typedef int          i32x4 __attribute__((ext_vector_type(4)));
typedef int          i32x8 __attribute__((ext_vector_type(8)));

union FragU { uint4 q[2]; v16h h; };
union H4    { _Float16 h[4]; uint2 u; };

__device__ __forceinline__ v8f wmma16(v16h a, v16h b, v8f c) {
  return __builtin_amdgcn_wmma_f32_16x16x32_f16(false, a, false, b, (short)0, c, false, false);
}

__device__ __forceinline__ float silu_f(float v) { return v / (1.0f + __expf(-v)); }

// Issue one TDM 2-D tile load: tile KC x tile_rows of 2-byte elems into LDS,
// padded +16B after each 128B row (LDS row stride 144B). D# per ISA ch.8.
__device__ __forceinline__ void tdm_load_2d(uint32_t lds_addr, const void* gptr,
                                            int tile_rows, int tensor_rows) {
  uint64_t ga = (uint64_t)(uintptr_t)gptr;
  u32x4 g0;
  g0.x = 1u;                                   // count=1 (valid), user mode
  g0.y = lds_addr;                             // bits 63:32  lds_addr
  g0.z = (uint32_t)ga;                         // bits 95:64  global_addr lo
  g0.w = (uint32_t)(ga >> 32) | 0x80000000u;   // addr hi + type=2 (bits 127:126)
  i32x8 g1;
  g1[0] = (1 << 16)        // data_size = 1 -> 2 bytes
        | (1 << 20)        // pad_enable
        | (4 << 22)        // pad_interval: 2^(4+1)=32 DWORDs = 128 B
        | (3 << 25);       // pad_amount: 3+1 = 4 DWORDs = 16 B
  g1[1] = (int)((DDIM & 0xFFFF) << 16);                 // tensor_dim0[15:0]
  g1[2] = (int)((DDIM >> 16) | ((tensor_rows & 0xFFFF) << 16)); // dim0 hi | dim1 lo
  g1[3] = (int)(((unsigned)tensor_rows >> 16) | (KC << 16));    // dim1 hi | tile_dim0
  g1[4] = tile_rows & 0xFFFF;                           // tile_dim1 ; tile_dim2=0
  g1[5] = DDIM;                                         // tensor_dim0_stride lo32
  g1[6] = 0;                                            // stride hi ; dim1_stride lo
  g1[7] = 0;
  i32x4 z4 = {0, 0, 0, 0};
  i32x8 z8 = {0, 0, 0, 0, 0, 0, 0, 0};
  __builtin_amdgcn_tensor_load_to_lds(g0, g1, z4, z4, z8, 0);
}

// ---------------------------------------------------------------------------
// Kernel 0: streaming f32 -> f16 conversion (x and gate[layer]) into workspace
// ---------------------------------------------------------------------------
__global__ __launch_bounds__(256) void cvt_f16(
    const float* __restrict__ src, _Float16* __restrict__ dst, int n4,
    const int* __restrict__ layer_idx, size_t lstride)
{
  size_t i = (size_t)blockIdx.x * 256 + threadIdx.x;
  if (i >= (size_t)n4) return;
  const float4 v = *(const float4*)(src + (size_t)layer_idx[0] * lstride + i * 4);
  H4 h;
  h.h[0] = (_Float16)v.x; h.h[1] = (_Float16)v.y;
  h.h[2] = (_Float16)v.z; h.h[3] = (_Float16)v.w;
  *(uint2*)(dst + i * 4) = h.u;
}

// ---------------------------------------------------------------------------
// Kernel 1: g_full[N,I] = silu( x[N,D] @ gate[layer][I,D]^T )  via f16 WMMA.
// Block tile 128 tok x 256 neu; 8 waves each 64x64 (4x4 frags, 32 wmma/chunk).
// Staging: TDM tensor_load_to_lds (TENSORcnt), double-buffered, KC=64.
// ---------------------------------------------------------------------------
__global__ __launch_bounds__(256) void gate_gemm_silu(
    const _Float16* __restrict__ xh, const _Float16* __restrict__ gh,
    float* __restrict__ gfull)
{
  extern __shared__ __align__(16) _Float16 smem[];
  _Float16* Ab = smem;                       // [2][AROWS][RSTR]
  _Float16* Bb = smem + 2 * AROWS * RSTR;    // [2][BROWS][RSTR]
#define AS_(b, r, c) Ab[((b) * AROWS + (r)) * RSTR + (c)]
#define BS_(b, r, c) Bb[((b) * BROWS + (r)) * RSTR + (c)]

  const int tid     = threadIdx.x;
  const int tokBase = blockIdx.y * 128;
  const int neuBase = blockIdx.x * 256;

  const int wave = tid >> 5;
  const int lane = tid & 31;
  const int half = lane >> 4;
  const int l16  = lane & 15;
  const int wm   = (wave & 1) * 64;
  const int wn   = (wave >> 1) * 64;

  const uint32_t aLds = (uint32_t)(uintptr_t)&AS_(0, 0, 0);
  const uint32_t bLds = (uint32_t)(uintptr_t)&BS_(0, 0, 0);
  const uint32_t aBufB = AROWS * RSTR * 2;   // bytes per A buffer
  const uint32_t bBufB = BROWS * RSTR * 2;   // bytes per B buffer

  v8f acc[4][4] = {};

  if (wave == 0) {
    tdm_load_2d(aLds, xh + (size_t)tokBase * DDIM, AROWS, NTOK);
    tdm_load_2d(bLds, gh + (size_t)neuBase * DDIM, BROWS, IDIM);
  }

  int buf = 0;
  for (int kc = 0; kc < DDIM; kc += KC) {
    if (wave == 0) __builtin_amdgcn_s_wait_tensorcnt(0);
    __syncthreads();
    if (wave == 0 && kc + KC < DDIM) {
      tdm_load_2d(aLds + (buf ^ 1) * aBufB,
                  xh + (size_t)tokBase * DDIM + kc + KC, AROWS, NTOK);
      tdm_load_2d(bLds + (buf ^ 1) * bBufB,
                  gh + (size_t)neuBase * DDIM + kc + KC, BROWS, IDIM);
    }

#pragma unroll
    for (int ks = 0; ks < 2; ks++) {
      const int k0 = ks * 32;
      // A frags: lanes 0-15 K=0..7/16..23 ; lanes 16-31 K=8..15/24..31
      FragU afr[4];
#pragma unroll
      for (int mf = 0; mf < 4; mf++) {
        int row = wm + mf * 16 + l16;
        afr[mf].q[0] = *(const uint4*)&AS_(buf, row, k0 + 8 * half);
        afr[mf].q[1] = *(const uint4*)&AS_(buf, row, k0 + 16 + 8 * half);
      }
      // B frags: lanes 0-15 K=0..7/16..23 ; lanes 16-31 K=8..15/24..31
#pragma unroll
      for (int nf = 0; nf < 4; nf++) {
        FragU bfr;
        int row = wn + nf * 16 + l16;
        bfr.q[0] = *(const uint4*)&BS_(buf, row, k0 + 8 * half);
        bfr.q[1] = *(const uint4*)&BS_(buf, row, k0 + 16 + 8 * half);
#pragma unroll
        for (int mf = 0; mf < 4; mf++)
          acc[mf][nf] = wmma16(afr[mf].h, bfr.h, acc[mf][nf]);
      }
    }
    buf ^= 1;
  }

  // Epilogue: C layout — VGPR r holds M = r + 8*half, N = lane&15
#pragma unroll
  for (int mf = 0; mf < 4; mf++) {
#pragma unroll
    for (int nf = 0; nf < 4; nf++) {
#pragma unroll
      for (int r = 0; r < 8; r++) {
        int tok = tokBase + wm + mf * 16 + r + 8 * half;
        int neu = neuBase + wn + nf * 16 + l16;
        gfull[(size_t)tok * IDIM + neu] = silu_f(acc[mf][nf][r]);
      }
    }
  }
}

// ---------------------------------------------------------------------------
// Kernel 2: per-token top-k by |silu|, then u_sel = dot(up[idx], x), gu.
// LDS fills use global_load_async_to_lds_b128 (ASYNCcnt).
// ---------------------------------------------------------------------------
__global__ __launch_bounds__(256) void topk_up_gu(
    const float* __restrict__ gfull, const float* __restrict__ x,
    const float* __restrict__ up, const int* __restrict__ layer_idx,
    const int* __restrict__ top_k, int* __restrict__ idxOut,
    float* __restrict__ guOut)
{
  __shared__ alignas(16) float vals[IDIM];
  __shared__ alignas(16) float xs[DDIM];
  __shared__ float red[256];
  __shared__ int   redi[256];
  __shared__ int   pidx[KMAX];
  __shared__ float pval[KMAX];

  const int n   = blockIdx.x;
  const int tid = threadIdx.x;
  const int kk  = min(top_k[0], KMAX);
  const int li  = layer_idx[0];

  {
    const uint32_t vB = (uint32_t)(uintptr_t)vals;
    const uint32_t xB = (uint32_t)(uintptr_t)xs;
    const char* gsrc = (const char*)(gfull + (size_t)n * IDIM);
    const char* xsrc = (const char*)(x + (size_t)n * DDIM);
#pragma unroll
    for (int it = 0; it < 8; it++) {            // 8192 f32 = 2048 x 16B
      uint32_t c = (uint32_t)(tid + it * 256) << 4;
      asm volatile("global_load_async_to_lds_b128 %0, %1, off"
                   :: "v"(vB + c), "v"((uint64_t)(uintptr_t)(gsrc + c)) : "memory");
    }
#pragma unroll
    for (int it = 0; it < 2; it++) {            // 2048 f32 = 512 x 16B
      uint32_t c = (uint32_t)(tid + it * 256) << 4;
      asm volatile("global_load_async_to_lds_b128 %0, %1, off"
                   :: "v"(xB + c), "v"((uint64_t)(uintptr_t)(xsrc + c)) : "memory");
    }
    asm volatile("s_wait_asynccnt 0x0" ::: "memory");
  }
  __syncthreads();

  for (int k = 0; k < kk; k++) {
    float bv = -1.0f; int bi = 0;
    for (int j = tid; j < IDIM; j += 256) {
      float a = fabsf(vals[j]);
      if (a > bv) { bv = a; bi = j; }
    }
    red[tid] = bv; redi[tid] = bi;
    __syncthreads();
    for (int s = 128; s > 0; s >>= 1) {
      if (tid < s && red[tid + s] > red[tid]) {
        red[tid] = red[tid + s]; redi[tid] = redi[tid + s];
      }
      __syncthreads();
    }
    if (tid == 0) {
      int j = redi[0];
      pidx[k] = j; pval[k] = vals[j]; vals[j] = 0.0f;
    }
    __syncthreads();
  }

  const float* __restrict__ uw = up + (size_t)li * IDIM * DDIM;
  for (int k = 0; k < kk; k++) {
    const float* __restrict__ urow = uw + (size_t)pidx[k] * DDIM;
    if (k + 1 < kk) __builtin_prefetch(uw + (size_t)pidx[k + 1] * DDIM + tid, 0, 0);
    float p = 0.0f;
    for (int d = tid; d < DDIM; d += 256) p += urow[d] * xs[d];
    red[tid] = p;
    __syncthreads();
    for (int s = 128; s > 0; s >>= 1) {
      if (tid < s) red[tid] += red[tid + s];
      __syncthreads();
    }
    if (tid == 0) {
      idxOut[(size_t)n * KMAX + k] = pidx[k];
      guOut [(size_t)n * KMAX + k] = pval[k] * red[0];
    }
    __syncthreads();
  }
}

// ---------------------------------------------------------------------------
// Kernel 3: dT[I,D] = down[layer][D,I]^T   (tiled 32x32 transpose)
// ---------------------------------------------------------------------------
__global__ __launch_bounds__(256) void transpose_down(
    const float* __restrict__ down, const int* __restrict__ layer_idx,
    float* __restrict__ dT)
{
  __shared__ float tile[32][33];
  const int li = layer_idx[0];
  const float* __restrict__ dw = down + (size_t)li * DDIM * IDIM;
  const int i0 = blockIdx.x * 32;
  const int d0 = blockIdx.y * 32;
  const int tx = threadIdx.x, ty = threadIdx.y;  // 32 x 8
#pragma unroll
  for (int s = 0; s < 32; s += 8)
    tile[ty + s][tx] = dw[(size_t)(d0 + ty + s) * IDIM + i0 + tx];
  __syncthreads();
#pragma unroll
  for (int s = 0; s < 32; s += 8)
    dT[(size_t)(i0 + ty + s) * DDIM + d0 + tx] = tile[tx][ty + s];
}

// ---------------------------------------------------------------------------
// Kernel 4: out[n,:] = sum_k gu[n,k] * dT[idx[n,k], :]   (coalesced rows)
// ---------------------------------------------------------------------------
__global__ __launch_bounds__(256) void down_proj(
    const float* __restrict__ dT, const int* __restrict__ idxIn,
    const float* __restrict__ guIn, const int* __restrict__ top_k,
    float* __restrict__ out)
{
  __shared__ int   sidx[KMAX];
  __shared__ float sgu[KMAX];
  const int n   = blockIdx.x;
  const int tid = threadIdx.x;
  const int kk  = min(top_k[0], KMAX);
  if (tid < kk) {
    sidx[tid] = idxIn[(size_t)n * KMAX + tid];
    sgu[tid]  = guIn [(size_t)n * KMAX + tid];
  }
  __syncthreads();

  float acc[DDIM / 256] = {};
  for (int k = 0; k < kk; k++) {
    const float* __restrict__ row = dT + (size_t)sidx[k] * DDIM;
    if (k + 1 < kk) __builtin_prefetch(dT + (size_t)sidx[k + 1] * DDIM + tid, 0, 0);
    const float g = sgu[k];
#pragma unroll
    for (int t = 0; t < DDIM / 256; t++) acc[t] += g * row[tid + t * 256];
  }
#pragma unroll
  for (int t = 0; t < DDIM / 256; t++)
    out[(size_t)n * DDIM + tid + t * 256] = acc[t];
}

// ---------------------------------------------------------------------------
extern "C" void kernel_launch(void* const* d_in, const int* in_sizes, int n_in,
                              void* d_out, int out_size, void* d_ws, size_t ws_size,
                              hipStream_t stream) {
  (void)in_sizes; (void)n_in; (void)out_size; (void)ws_size;

  const float* x     = (const float*)d_in[0];
  const float* gate  = (const float*)d_in[1];
  const float* up    = (const float*)d_in[2];
  const float* down  = (const float*)d_in[3];
  const int* layer_i = (const int*)d_in[4];
  const int* top_k   = (const int*)d_in[5];
  float* out = (float*)d_out;

  char* ws = (char*)d_ws;
  float*     gfull = (float*)ws;                                   // N*I f32  (64 MB)
  char*      p1    = ws + (size_t)NTOK * IDIM * sizeof(float);
  float*     dT    = (float*)p1;                                   // I*D f32  (64 MB)
  char*      p2    = p1 + (size_t)IDIM * DDIM * sizeof(float);
  _Float16*  xh    = (_Float16*)p2;                                // N*D f16  (8 MB)
  char*      p3    = p2 + (size_t)NTOK * DDIM * sizeof(_Float16);
  _Float16*  gh    = (_Float16*)p3;                                // I*D f16  (32 MB)
  char*      p4    = p3 + (size_t)IDIM * DDIM * sizeof(_Float16);
  int*       idx   = (int*)p4;                                     // N*KMAX
  char*      p5    = p4 + (size_t)NTOK * KMAX * sizeof(int);
  float*     gu    = (float*)p5;                                   // N*KMAX

  const int n4x = NTOK * DDIM / 4;
  const int n4g = IDIM * DDIM / 4;
  cvt_f16<<<(n4x + 255) / 256, 256, 0, stream>>>(x, xh, n4x, layer_i, 0);
  cvt_f16<<<(n4g + 255) / 256, 256, 0, stream>>>(gate, gh, n4g, layer_i,
                                                 (size_t)IDIM * DDIM);

  const size_t smemB = (size_t)(2 * AROWS * RSTR + 2 * BROWS * RSTR) * sizeof(_Float16);
  gate_gemm_silu<<<dim3(IDIM / 256, NTOK / 128), 256, smemB, stream>>>(xh, gh, gfull);
  transpose_down<<<dim3(IDIM / 32, DDIM / 32), dim3(32, 8), 0, stream>>>(down, layer_i, dT);
  topk_up_gu<<<NTOK, 256, 0, stream>>>(gfull, x, up, layer_i, top_k, idx, gu);
  down_proj<<<NTOK, 256, 0, stream>>>(dT, idx, gu, top_k, out);
}
